// AttentionCell_5557687681534
// MI455X (gfx1250) — matrix operand order repaired and checked by reading, not compiled, not run on hardware
//
#include <hip/hip_runtime.h>

// ---------------------------------------------------------------------------
// MI455X (gfx1250) attention cell.
// Compute-bound (~103 GFLOP vs ~100MB traffic @23.3TB/s) -> all matmuls run
// through v_wmma_f32_16x16x32_f16 (f16 in, fp32 accumulate), wave32 layouts
// per CDNA5 ISA 7.12.2. Flash-style online softmax avoids the 1GB score
// tensor. K/V tiles are staged with GLOBAL_LOAD_ASYNC_TO_LDS_B128 (ASYNCcnt)
// and double-buffered so the async engine overlaps the WMMA pipeline.
// ---------------------------------------------------------------------------

typedef __attribute__((ext_vector_type(16))) _Float16 v16h;
typedef __attribute__((ext_vector_type(8)))  _Float16 v8h;
typedef __attribute__((ext_vector_type(4)))  _Float16 v4h;
typedef __attribute__((ext_vector_type(8)))  float    v8f;
typedef _Float16 h16;

#define S_LEN 4096
#define MODEL_DIM 1024
#define N_HEADS 16
#define HEAD_DIM 64

// ---- CDNA5 async global->LDS (ASYNCcnt-tracked), per ISA 10. / 15.18.3 ----
// Generic pointers to __shared__ carry the LDS offset in their low 32 bits
// (flat shared-aperture encoding), which matches VDST semantics:
//   the VDST VGPR gives the LDS byte address (plus the instruction offset).
__device__ __forceinline__ void async_load_lds_b128(void* lds_ptr, const void* gptr) {
    unsigned int loff = (unsigned int)(unsigned long long)lds_ptr;
    asm volatile("global_load_async_to_lds_b128 %0, %1, off"
                 :: "v"(loff), "v"(gptr)
                 : "memory");
}
__device__ __forceinline__ void wait_asynccnt0() {
    asm volatile("s_wait_asynccnt 0x0" ::: "memory");
}

// Pack a 16-half fragment from two contiguous 8-half (16B-aligned) groups.
// Used for both A and B fragments: K kb..kb+15 contiguous per half-wave
// (this code's shared k order, not AMD's table order).
__device__ __forceinline__ v16h pack_ab(const h16* p_lo, const h16* p_hi) {
    v8h lo = *(const v8h*)p_lo;
    v8h hi = *(const v8h*)p_hi;
    v16h r;
#pragma unroll
    for (int i = 0; i < 8; ++i) { r[i] = lo[i]; r[8 + i] = hi[i]; }
    return r;
}

// ---------------------------------------------------------------------------
// WMMA GEMM: C(MxN, fp32) = A(MxK) * B(KxN, fp32) + bias.
// A is fp32 (A_F16=false, converted in LDS) or f16 (A_F16=true, async-staged).
// 256 threads = 8 waves. Block tile 128x128, K-step 32, wave tile 64x32.
// ---------------------------------------------------------------------------
template <bool A_F16>
__global__ __launch_bounds__(256)
void wmma_gemm_kernel(const void* __restrict__ Aq, const float* __restrict__ B,
                      const float* __restrict__ bias, float* __restrict__ C,
                      int M, int N, int K)
{
    __shared__ alignas(16) h16 As[128][40];   // [m][k], +8 pad keeps 16B rows
    __shared__ alignas(16) h16 Bs[128][40];   // transposed: [n][k]

    const int tid   = threadIdx.x;
    const int lane  = tid & 31;
    const int w     = tid >> 5;
    const int lk    = lane & 15;          // N (or M) index within a 16-tile
    const int halfv = lane >> 4;          // half-wave select
    const int kg    = halfv * 8;          // unused (A fragments also use kb)
    const int kb    = halfv * 16;         // B-fragment K base
    const int m0    = blockIdx.y * 128;
    const int n0    = blockIdx.x * 128;
    const int mbase = (w >> 2) * 64;
    const int nbase = (w & 3) * 32;

    v8f acc[4][2] = {};

    for (int k0 = 0; k0 < K; k0 += 32) {
        __syncthreads();
        if (A_F16) {
            const h16* A16 = (const h16*)Aq;
#pragma unroll
            for (int q = 0; q < 2; ++q) {
                int g = tid + q * 256;
                int row = g >> 2, seg = g & 3;
                async_load_lds_b128(&As[row][seg * 8],
                                    A16 + (size_t)(m0 + row) * K + k0 + seg * 8);
            }
        } else {
            const float* A32 = (const float*)Aq;
#pragma unroll
            for (int q = 0; q < 4; ++q) {
                int f = tid + q * 256;
                int row = f >> 3, c4 = f & 7;
                const float4 v = *(const float4*)(A32 + (size_t)(m0 + row) * K + k0 + c4 * 4);
                v4h hv; hv[0] = (h16)v.x; hv[1] = (h16)v.y; hv[2] = (h16)v.z; hv[3] = (h16)v.w;
                *(v4h*)&As[row][c4 * 4] = hv;
            }
            if (k0 + 32 < K) {  // global_prefetch_b8 for next A tile
                int row = tid >> 3, c4 = tid & 7;
                __builtin_prefetch(A32 + (size_t)(m0 + row) * K + k0 + 32 + c4 * 4, 0, 1);
            }
        }
        // B tile, stored transposed so B-fragments are contiguous in K.
#pragma unroll
        for (int q = 0; q < 4; ++q) {
            int f = tid + q * 256;
            int kr = f >> 5, c4 = f & 31;
            const float4 v = *(const float4*)(B + (size_t)(k0 + kr) * N + n0 + c4 * 4);
            Bs[c4 * 4 + 0][kr] = (h16)v.x;
            Bs[c4 * 4 + 1][kr] = (h16)v.y;
            Bs[c4 * 4 + 2][kr] = (h16)v.z;
            Bs[c4 * 4 + 3][kr] = (h16)v.w;
        }
        if (k0 + 32 < K) {      // global_prefetch_b8 for next B tile
            int kr = tid >> 5, c4 = tid & 31;
            __builtin_prefetch(B + (size_t)(k0 + 32 + kr) * N + n0 + c4 * 4, 0, 1);
        }
        if (A_F16) wait_asynccnt0();
        __syncthreads();

        v16h bfr[2];
#pragma unroll
        for (int j = 0; j < 2; ++j) {
            int col = nbase + j * 16 + lk;
            bfr[j] = pack_ab(&Bs[col][kb], &Bs[col][kb + 8]);
        }
#pragma unroll
        for (int i = 0; i < 4; ++i) {
            int row = mbase + i * 16 + lk;
            v16h afr = pack_ab(&As[row][kb], &As[row][kb + 8]);
#pragma unroll
            for (int j = 0; j < 2; ++j)
                acc[i][j] = __builtin_amdgcn_wmma_f32_16x16x32_f16(
                    false, afr, false, bfr[j], (short)0, acc[i][j], false, false);
        }
    }

    // Epilogue: C layout rows M = r + 8*halfv, cols N = lk.
#pragma unroll
    for (int j = 0; j < 2; ++j) {
        int col = n0 + nbase + j * 16 + lk;
        float bv = bias[col];
#pragma unroll
        for (int i = 0; i < 4; ++i) {
#pragma unroll
            for (int r = 0; r < 8; ++r) {
                int row = m0 + mbase + i * 16 + r + 8 * halfv;
                C[(size_t)row * N + col] = acc[i][j][r] + bv;
            }
        }
    }
}

// ---------------------------------------------------------------------------
// RoPE (full model dim, rotate-half pairs i / i+512) + pack to head-major f16.
// Folds the 1/sqrt(HEAD_DIM)=0.125 score scale into Q.
// ---------------------------------------------------------------------------
__global__ __launch_bounds__(512)
void rope_pack_kernel(const float* __restrict__ qkv, h16* __restrict__ Qh,
                      h16* __restrict__ Kh, h16* __restrict__ Vh)
{
    const int s = blockIdx.x;
    const int i = threadIdx.x;  // 0..511, pair (i, i+512)
    // inv_freq = 10000^(-2i/1024); log2(10000) = 13.2877123795...
    const float inv_freq = exp2f(-((float)(2 * i) * (1.0f / 1024.0f)) * 13.2877123795494f);
    const float ang = (float)s * inv_freq;
    float sn, cs;
    __sincosf(ang, &sn, &cs);

    const float* row = qkv + (size_t)s * 3072;
    float q0 = row[i],        q1 = row[i + 512];
    float k0 = row[1024 + i], k1 = row[1024 + i + 512];
    float v0 = row[2048 + i], v1 = row[2048 + i + 512];
    float qo0 = (q0 * cs - q1 * sn) * 0.125f;
    float qo1 = (q1 * cs + q0 * sn) * 0.125f;
    float ko0 = k0 * cs - k1 * sn;
    float ko1 = k1 * cs + k0 * sn;

    int h0 = i >> 6, h1 = (i + 512) >> 6, dh = i & 63;  // (i+512)%64 == i%64
    size_t o0 = ((size_t)h0 * S_LEN + s) * HEAD_DIM + dh;
    size_t o1 = ((size_t)h1 * S_LEN + s) * HEAD_DIM + dh;
    Qh[o0] = (h16)qo0; Qh[o1] = (h16)qo1;
    Kh[o0] = (h16)ko0; Kh[o1] = (h16)ko1;
    Vh[o0] = (h16)v0;  Vh[o1] = (h16)v1;
}

// ---------------------------------------------------------------------------
// Flash attention: grid (32 q-blocks, 16 heads), 256 threads = 8 waves.
// Wave w owns q rows [qbase+16w, qbase+16w+15]; Q fragments live in regs.
// K/V tiles (32 kv x 64) staged via async-to-LDS b128 and double-buffered:
// chunk c+1 loads are issued right after the s_wait_asynccnt/barrier that
// publishes chunk c, so DMA overlaps the 8 WMMAs + softmax of chunk c.
// ---------------------------------------------------------------------------
__global__ __launch_bounds__(256)
void flash_attn_kernel(const h16* __restrict__ Qh, const h16* __restrict__ Kh,
                       const h16* __restrict__ Vh, h16* __restrict__ attn_out)
{
    __shared__ alignas(16) h16 Ks[2][32][72];  // [buf][kv_local][k(64)+pad]
    __shared__ alignas(16) h16 Vs[2][32][72];  // [buf][kv_local][d(64)+pad]
    __shared__ alignas(16) h16 Ps[8][16][40];  // per-wave P tile [row][32 kv]+pad

    const int tid   = threadIdx.x;
    const int lane  = tid & 31;
    const int w     = tid >> 5;
    const int lk    = lane & 15;
    const int halfv = lane >> 4;
    const int kg    = halfv * 8;
    const int kb    = halfv * 16;
    const int h     = blockIdx.y;
    const int qbase = blockIdx.x * 128;
    const int qrow0 = qbase + w * 16;

    const h16* Qhead = Qh + (size_t)h * S_LEN * HEAD_DIM;
    const h16* Khead = Kh + (size_t)h * S_LEN * HEAD_DIM;
    const h16* Vhead = Vh + (size_t)h * S_LEN * HEAD_DIM;

    // staging coordinates (one async b128 per thread per tile)
    const int ksl = tid >> 3, kseg = tid & 7;       // K tile: [sl][seg*8]
    const int vsl = tid & 31, vd0 = (tid >> 5) * 8; // V tile: [sl][d0]

    // Q A-fragments (16 rows x 64 head-dim = two K=32 fragments), regs-resident.
    v16h qf[2];
#pragma unroll
    for (int kf = 0; kf < 2; ++kf) {
        const h16* p = Qhead + (size_t)(qrow0 + lk) * HEAD_DIM + kf * 32;
        qf[kf] = pack_ab(p + kb, p + kb + 8);
    }

    v8f acc[4] = {};
    float mrow[8], lrow[8];
#pragma unroll
    for (int r = 0; r < 8; ++r) { mrow[r] = -1e30f; lrow[r] = 0.0f; }

    const int nchunks = (qbase + 128) >> 5;

    // prologue: stage chunk 0 into buffer 0
    async_load_lds_b128(&Ks[0][ksl][kseg * 8], Khead + (size_t)ksl * HEAD_DIM + kseg * 8);
    async_load_lds_b128(&Vs[0][vsl][vd0],      Vhead + (size_t)vsl * HEAD_DIM + vd0);

    for (int c = 0; c < nchunks; ++c) {
        const int kvb = c << 5;
        const int buf = c & 1;
        wait_asynccnt0();     // chunk c resident in LDS (this wave's asyncs done)
        __syncthreads();      // ...and published by every wave

        if (c + 1 < nchunks) {  // kick chunk c+1 into the other buffer
            const int kvn = kvb + 32;
            async_load_lds_b128(&Ks[buf ^ 1][ksl][kseg * 8],
                                Khead + (size_t)(kvn + ksl) * HEAD_DIM + kseg * 8);
            async_load_lds_b128(&Vs[buf ^ 1][vsl][vd0],
                                Vhead + (size_t)(kvn + vsl) * HEAD_DIM + vd0);
        }

        if (kvb <= qrow0 + 15) {   // wave-uniform: EXEC stays all-1s inside
            // scores: two 16x16 tiles over K=64 (2 x K=32 wmma each)
            v8f sc[2] = {};
#pragma unroll
            for (int t = 0; t < 2; ++t) {
                int col = t * 16 + lk;
#pragma unroll
                for (int kf = 0; kf < 2; ++kf) {
                    v16h bf = pack_ab(&Ks[buf][col][kf * 32 + kb],
                                      &Ks[buf][col][kf * 32 + kb + 8]);
                    sc[t] = __builtin_amdgcn_wmma_f32_16x16x32_f16(
                        false, qf[kf], false, bf, (short)0, sc[t], false, false);
                }
            }
            // online softmax (rows of a 16-tile live across one half-wave)
#pragma unroll
            for (int r = 0; r < 8; ++r) {
                const int qg = qrow0 + r + 8 * halfv;
#pragma unroll
                for (int t = 0; t < 2; ++t) {
                    int kvg = kvb + t * 16 + lk;
                    if (kvg > qg) sc[t][r] = -1e30f;   // causal mask
                }
                float mx = fmaxf(sc[0][r], sc[1][r]);
#pragma unroll
                for (int o = 8; o > 0; o >>= 1) mx = fmaxf(mx, __shfl_xor(mx, o, 16));
                float mn    = fmaxf(mrow[r], mx);
                float alpha = __expf(mrow[r] - mn);
                float p0    = __expf(sc[0][r] - mn);
                float p1    = __expf(sc[1][r] - mn);
                float sum   = p0 + p1;
#pragma unroll
                for (int o = 8; o > 0; o >>= 1) sum += __shfl_xor(sum, o, 16);
                lrow[r] = lrow[r] * alpha + sum;
                mrow[r] = mn;
#pragma unroll
                for (int u = 0; u < 4; ++u) acc[u][r] *= alpha;
                // stash P (C layout -> row major) in wave-private LDS
                Ps[w][r + 8 * halfv][lk]      = (h16)p0;
                Ps[w][r + 8 * halfv][16 + lk] = (h16)p1;
            }
            // P(16x32) x V(32x64): reload P as A-fragment; V B-fragments are
            // column reads of the row-major async-staged tile.
            v16h pf = pack_ab(&Ps[w][lk][kb], &Ps[w][lk][kb + 8]);
#pragma unroll
            for (int u = 0; u < 4; ++u) {
                const h16* vp = &Vs[buf][kb][u * 16 + lk];
                v16h bf;
#pragma unroll
                for (int i = 0; i < 16; ++i) bf[i] = vp[(size_t)i * 72];
                acc[u] = __builtin_amdgcn_wmma_f32_16x16x32_f16(
                    false, pf, false, bf, (short)0, acc[u], false, false);
            }
        }
    }

    // normalize and store f16 at [s][h*64 + d] (input layout for output GEMM)
#pragma unroll
    for (int r = 0; r < 8; ++r) {
        float inv = 1.0f / lrow[r];
        int qg = qrow0 + r + 8 * halfv;
        h16* orow = attn_out + (size_t)qg * MODEL_DIM + h * HEAD_DIM;
#pragma unroll
        for (int u = 0; u < 4; ++u)
            orow[u * 16 + lk] = (h16)(acc[u][r] * inv);
    }
}

// ---------------------------------------------------------------------------
// LayerNorm over dim=1024, one block per row.
// ---------------------------------------------------------------------------
__global__ __launch_bounds__(256)
void layernorm_kernel(const float* __restrict__ X, const float* __restrict__ gamma,
                      const float* __restrict__ beta, float* __restrict__ out)
{
    const int row = blockIdx.x;
    const int tid = threadIdx.x;
    const float* x = X + (size_t)row * MODEL_DIM;
    float s = 0.f, ss = 0.f;
#pragma unroll
    for (int i = tid; i < MODEL_DIM; i += 256) { float v = x[i]; s += v; ss += v * v; }
#pragma unroll
    for (int o = 16; o > 0; o >>= 1) { s += __shfl_xor(s, o, 32); ss += __shfl_xor(ss, o, 32); }
    __shared__ float sh[2][8];
    if ((tid & 31) == 0) { sh[0][tid >> 5] = s; sh[1][tid >> 5] = ss; }
    __syncthreads();
    s = 0.f; ss = 0.f;
#pragma unroll
    for (int q = 0; q < 8; ++q) { s += sh[0][q]; ss += sh[1][q]; }
    const float mu   = s * (1.0f / MODEL_DIM);
    const float var  = ss * (1.0f / MODEL_DIM) - mu * mu;
    const float rstd = rsqrtf(var + 1e-5f);
    float* o = out + (size_t)row * MODEL_DIM;
    for (int i = tid; i < MODEL_DIM; i += 256)
        o[i] = (x[i] - mu) * rstd * gamma[i] + beta[i];
}

// ---------------------------------------------------------------------------
// Launch. Workspace layout (bytes):
//   [0, 48MiB)      qkv fp32 (dead after rope; proj fp32 16MiB aliases it)
//   [48MiB, 56MiB)  Qh f16   [56,64) Kh   [64,72) Vh   [72,80) attn f16
// ---------------------------------------------------------------------------
extern "C" void kernel_launch(void* const* d_in, const int* in_sizes, int n_in,
                              void* d_out, int out_size, void* d_ws, size_t ws_size,
                              hipStream_t stream)
{
    const float* x     = (const float*)d_in[0];
    const float* Wqkv  = (const float*)d_in[1];
    const float* bqkv  = (const float*)d_in[2];
    const float* Wo    = (const float*)d_in[3];
    const float* bo    = (const float*)d_in[4];
    const float* gamma = (const float*)d_in[5];
    const float* beta  = (const float*)d_in[6];

    char* ws = (char*)d_ws;
    float* qkv  = (float*)ws;                              // 48 MiB
    float* proj = (float*)ws;                              // 16 MiB (aliases dead qkv)
    h16* Qh     = (h16*)(ws + (size_t)50331648);           // 8 MiB
    h16* Kh     = (h16*)(ws + (size_t)58720256);           // 8 MiB
    h16* Vh     = (h16*)(ws + (size_t)67108864);           // 8 MiB
    h16* attn   = (h16*)(ws + (size_t)75497472);           // 8 MiB

    wmma_gemm_kernel<false><<<dim3(24, 32), 256, 0, stream>>>(x, Wqkv, bqkv, qkv,
                                                              S_LEN, 3 * MODEL_DIM, MODEL_DIM);
    rope_pack_kernel<<<dim3(S_LEN), 512, 0, stream>>>(qkv, Qh, Kh, Vh);
    flash_attn_kernel<<<dim3(32, N_HEADS), 256, 0, stream>>>(Qh, Kh, Vh, attn);
    wmma_gemm_kernel<true><<<dim3(8, 32), 256, 0, stream>>>(attn, Wo, bo, proj,
                                                            S_LEN, MODEL_DIM, MODEL_DIM);
    layernorm_kernel<<<dim3(S_LEN), 256, 0, stream>>>(proj, gamma, beta, (float*)d_out);
}
